// SoftmaxTransformerBlock_52707838656611
// MI455X (gfx1250) — matrix operand order repaired and checked by reading, not compiled, not run on hardware
//
#include <hip/hip_runtime.h>
#include <hip/hip_bf16.h>

// ---------------------------------------------------------------------------
// Transformer block on gfx1250 (MI455X).
// - All GEMMs + flash-attention use v_wmma_f32_16x16x32_bf16 (f32 accumulate).
// - GEMM B-operand staged by the Tensor Data Mover (tensor_load_to_lds,
//   double-buffered, s_wait_tensorcnt) when the builtin exists; manual LDS
//   copy fallback otherwise.
// ---------------------------------------------------------------------------

#define DMODEL 1024
#define DFFN   4096
#define NHEADS 16
#define DHEAD  64
#define BB     2
#define TT     2048
#define MTOK   (BB * TT)   // 4096 token rows

using u32 = unsigned int;
using u16 = unsigned short;

typedef __attribute__((ext_vector_type(16))) __bf16 v16bf;
typedef __attribute__((ext_vector_type(8)))  float  v8f;
typedef __attribute__((ext_vector_type(8)))  u32    v8u;
typedef __attribute__((ext_vector_type(4)))  u32    v4u;

struct U8pair { v4u lo, hi; };

#if defined(__HIP_DEVICE_COMPILE__) && __has_builtin(__builtin_amdgcn_tensor_load_to_lds)
#define USE_TDM 1
#else
#define USE_TDM 0
#endif

// ---- helpers --------------------------------------------------------------

__device__ __forceinline__ u16 f2bf(float f) {
  u32 x = __float_as_uint(f);
  x += 0x7fffu + ((x >> 16) & 1u);   // round-to-nearest-even
  return (u16)(x >> 16);
}
__device__ __forceinline__ u32 packbf(float a, float b) {
  return (u32)f2bf(a) | ((u32)f2bf(b) << 16);
}

// A-fragment (16-bit A 16x32): V0..3 at p[0..3], V4..7 at p[8..11]
__device__ __forceinline__ v16bf ldfragA(const u32* p) {
  U8pair t;
  t.lo = *(const v4u*)p;
  t.hi = *(const v4u*)(p + 8);
  return __builtin_bit_cast(v16bf, t);
}
// B-fragment (table order, same as A): V0..3 at p[0..3], V4..7 at p[8..11]
__device__ __forceinline__ v16bf ldfragB(const u32* p) {
  U8pair t;
  t.lo = *(const v4u*)p;
  t.hi = *(const v4u*)(p + 8);
  return __builtin_bit_cast(v16bf, t);
}
__device__ __forceinline__ v8f wmma_bf16(v16bf a, v16bf b, v8f c) {
  return __builtin_amdgcn_wmma_f32_16x16x32_bf16(
      /*neg_a=*/false, a, /*neg_b=*/false, b,
      /*c_mod=*/(short)0, c, /*reuse_a=*/false, /*reuse_b=*/false);
}
__device__ __forceinline__ v8f vzero8() {
  v8f r;
#pragma unroll
  for (int i = 0; i < 8; ++i) r[i] = 0.f;
  return r;
}

// ---- Tensor Data Mover: 2D tile (tile_k contiguous elems x tile_n rows) ---
// of a bf16 tensor with row stride K_elems, into LDS at byte offset lds_off.
// D# per CDNA5 ISA (group0: count=1,type=2; group1:
// data_size=2B, dims/strides packed into their bitfields).

#if USE_TDM
typedef __attribute__((ext_vector_type(4))) unsigned int tdm_u32x4;
typedef __attribute__((ext_vector_type(8))) int          tdm_i32x8;
typedef __attribute__((ext_vector_type(4))) int          tdm_i32x4;

__device__ __forceinline__ void tdm_load_tile(u32 lds_off, const u16* g,
                                              u32 K_elems, u32 n_rows,
                                              u32 tile_k, u32 tile_n) {
  unsigned long long ga = (unsigned long long)(size_t)g;
  tdm_u32x4 g0 = { 1u,                                  // count=1 (valid D#)
                   lds_off,                             // lds_addr
                   (u32)ga,                             // global_addr[31:0]
                   (u32)(ga >> 32) | (2u << 30) };      // ga[56:32] | type=2
  tdm_i32x8 g1 = { (int)(1u << 16),                     // data_size=1 (2 bytes)
                   (int)(K_elems << 16),                // tensor_dim0[15:0]
                   (int)((K_elems >> 16) | (n_rows << 16)),   // td0 hi | td1 lo
                   (int)((n_rows >> 16) | (tile_k << 16)),    // td1 hi | tile_dim0
                   (int)tile_n,                         // tile_dim1 (tile_dim2=0)
                   (int)K_elems,                        // tensor_dim0_stride lo32
                   0, 0 };
  tdm_i32x4 gz = { 0, 0, 0, 0 };
#if __has_include(<hip/amd_detail/amd_gfx1250_TDM.h>)
  tdm_i32x8 gz8 = { 0, 0, 0, 0, 0, 0, 0, 0 };
  __builtin_amdgcn_tensor_load_to_lds(g0, g1, gz, gz, gz8, 0);   // clang-23 form
#else
  __builtin_amdgcn_tensor_load_to_lds(g0, g1, gz, gz, 0);        // ROCm 7.2 form
#endif
}
#endif

// ---- weight convert + transpose: f32 W[K][N] -> bf16 Wt[N][K] -------------

__global__ void k_transpose_bf16(const float* __restrict__ in,
                                 u16* __restrict__ out, int K, int N) {
  long long idx = (long long)blockIdx.x * blockDim.x + threadIdx.x;
  long long total = (long long)K * N;
  if (idx >= total) return;
  int k = (int)(idx % K);
  int n = (int)(idx / K);
  out[idx] = f2bf(in[(long long)k * N + n]);   // writes coalesced
}

// ---- layernorm (f32 in -> bf16 out), one row (1024) per 256-thread block --

__global__ __launch_bounds__(256) void k_layernorm_bf16(
    const float* __restrict__ x, const float* __restrict__ w,
    const float* __restrict__ b, u16* __restrict__ out) {
  const int row = blockIdx.x, tid = threadIdx.x;
  __shared__ float red[8];
  const float* xr = x + (size_t)row * DMODEL;
  float4 v = *(const float4*)(xr + tid * 4);
  float s = v.x + v.y + v.z + v.w;
#pragma unroll
  for (int off = 16; off >= 1; off >>= 1) s += __shfl_xor(s, off, 32);
  if ((tid & 31) == 0) red[tid >> 5] = s;
  __syncthreads();
  float tot = 0.f;
#pragma unroll
  for (int i = 0; i < 8; ++i) tot += red[i];
  const float mean = tot * (1.f / DMODEL);
  __syncthreads();
  float d0 = v.x - mean, d1 = v.y - mean, d2 = v.z - mean, d3 = v.w - mean;
  float s2 = d0 * d0 + d1 * d1 + d2 * d2 + d3 * d3;
#pragma unroll
  for (int off = 16; off >= 1; off >>= 1) s2 += __shfl_xor(s2, off, 32);
  if ((tid & 31) == 0) red[tid >> 5] = s2;
  __syncthreads();
  float tv = 0.f;
#pragma unroll
  for (int i = 0; i < 8; ++i) tv += red[i];
  const float rstd = rsqrtf(tv * (1.f / DMODEL) + 1e-5f);
  u16* orow = out + (size_t)row * DMODEL;
  float vv[4] = {v.x, v.y, v.z, v.w};
#pragma unroll
  for (int j = 0; j < 4; ++j) {
    int c = tid * 4 + j;
    orow[c] = f2bf((vv[j] - mean) * rstd * w[c] + b[c]);
  }
}

// ---- generic bf16 WMMA GEMM: C = alpha*(A @ Bt^T) (+res) (GELU?) ----------
// A: [M][K] bf16 row-major. Bt: [N][K] bf16 (i.e. B transposed).
// Block tile 64(M) x 256(N), 8 waves (2x4) of 32x64, K-step 32.
// B tiles staged by TDM (double-buffered, prefetch one step ahead).

template <bool GELU>
__global__ __launch_bounds__(256) void k_gemm_bf16(
    const u16* __restrict__ A, const u16* __restrict__ Bt,
    int M, int N, int K, float alpha,
    const float* __restrict__ res, float* __restrict__ outF,
    u16* __restrict__ outB) {
  __shared__ __attribute__((aligned(16))) u32 As[64 * 16];       // 64 x 32 bf16
  __shared__ __attribute__((aligned(16))) u32 Bs[2][256 * 16];   // 2x 256 x 32 bf16
  const int tid = threadIdx.x;
  const int lane = tid & 31, wv = tid >> 5;
  const int wm = wv & 1, wn = wv >> 1;      // 2 x 4 wave grid
  const int l15 = lane & 15, hi = lane >> 4;
  const int mblk = blockIdx.y * 64, nblk = blockIdx.x * 256;

  v8f acc[2][4];
#pragma unroll
  for (int mt = 0; mt < 2; ++mt)
#pragma unroll
    for (int nt = 0; nt < 4; ++nt) acc[mt][nt] = vzero8();

  const u32* Au = (const u32*)A;
  const int Ku = K >> 1;        // uints per row
  const int steps = K >> 5;     // 32-K steps

  for (int s = 0; s < steps; ++s) {
    const int k0 = s << 5;
    __syncthreads();            // previous step's readers are done
    {  // stage A tile: 64 rows x 16 uints, one 16B copy per thread
      int row = tid >> 2, seg = tid & 3;
      *(v4u*)(As + row * 16 + seg * 4) =
          *(const v4u*)(Au + (size_t)(mblk + row) * Ku + (k0 >> 1) + seg * 4);
    }
#if USE_TDM
    if (wv == 0) {   // wave 0 drives the TDM pipeline (EXEC ignored by TDM)
      if (s == 0)
        tdm_load_tile((u32)(size_t)&Bs[0][0],
                      Bt + (size_t)nblk * K, (u32)K, 256u, 32u, 256u);
      if (s + 1 < steps) {
        tdm_load_tile((u32)(size_t)&Bs[(s + 1) & 1][0],
                      Bt + (size_t)nblk * K + (k0 + 32), (u32)K, 256u, 32u, 256u);
        __builtin_amdgcn_s_wait_tensorcnt((short)1);   // tile s complete
      } else {
        __builtin_amdgcn_s_wait_tensorcnt((short)0);
      }
    }
#else
    {  // fallback: manual B tile copy (256 rows x 16 uints)
      u32* Bb = &Bs[s & 1][0];
#pragma unroll
      for (int j = 0; j < 4; ++j) {
        int i = tid + j * 256;
        int row = i >> 2, seg = i & 3;
        *(v4u*)(Bb + row * 16 + seg * 4) =
            *(const v4u*)((const u32*)Bt + (size_t)(nblk + row) * Ku +
                          (k0 >> 1) + seg * 4);
      }
    }
#endif
    __syncthreads();            // A staged + B tile s visible to all waves

    const u32* Bb = &Bs[s & 1][0];
    v16bf af[2], bfr[4];
#pragma unroll
    for (int mt = 0; mt < 2; ++mt)
      af[mt] = ldfragA(As + (wm * 32 + mt * 16 + l15) * 16 + hi * 4);
#pragma unroll
    for (int nt = 0; nt < 4; ++nt)
      bfr[nt] = ldfragB(Bb + (wn * 64 + nt * 16 + l15) * 16 + hi * 4);
#pragma unroll
    for (int mt = 0; mt < 2; ++mt)
#pragma unroll
      for (int nt = 0; nt < 4; ++nt)
        acc[mt][nt] = wmma_bf16(af[mt], bfr[nt], acc[mt][nt]);
  }

#pragma unroll
  for (int mt = 0; mt < 2; ++mt)
#pragma unroll
    for (int nt = 0; nt < 4; ++nt) {
#pragma unroll
      for (int i = 0; i < 8; ++i) {
        int row = mblk + wm * 32 + mt * 16 + hi * 8 + i;
        int col = nblk + wn * 64 + nt * 16 + l15;
        size_t off = (size_t)row * N + col;
        float v = acc[mt][nt][i] * alpha;
        if (res) v += res[off];
        if (GELU) v = 0.5f * v * (1.f + erff(v * 0.70710678118654752f));
        if (outF) outF[off] = v;
        if (outB) outB[off] = f2bf(v);
      }
    }
}

// ---- causal flash attention -----------------------------------------------
// Scores computed transposed (S^T = K_blk @ Q^T) so exp(S^T) in C-layout is
// exactly the A-fragment of P for the P@V WMMA. One block = 128 q rows
// (8 waves x 16); K and V^T staged in LDS per 32-k block. Causal masking is
// applied only on diagonal blocks (wave-uniform branch).

__global__ __launch_bounds__(256) void k_attention(
    const u16* __restrict__ Qg, const u16* __restrict__ Kg,
    const u16* __restrict__ Vg, u16* __restrict__ Og) {
  __shared__ __attribute__((aligned(16))) u32 Ks[32 * 32];   // 32 k x 64 d bf16
  __shared__ __attribute__((aligned(16))) u16 Vs[64 * 32];   // 64 dv x 32 k
  const int tid = threadIdx.x, lane = tid & 31, wv = tid >> 5;
  const int l15 = lane & 15, hi = lane >> 4;
  const int qblk = blockIdx.x;
  const int b = blockIdx.y >> 4, h = blockIdx.y & 15;
  const u32* Qu = (const u32*)Qg + (size_t)b * TT * 512 + h * 32;
  const u32* Ku = (const u32*)Kg + (size_t)b * TT * 512 + h * 32;
  const u32* Vu = (const u32*)Vg + (size_t)b * TT * 512 + h * 32;
  const int qw = qblk * 128 + wv * 16;   // wave's first q row
  const int q = qw + l15;                // this lane's q (stats + B-frag col)

  v16bf qfrag[2];   // Q^T B-fragments, d = 0..31 / 32..63 (Q pre-scaled 1/8)
#pragma unroll
  for (int d = 0; d < 2; ++d)
    qfrag[d] = ldfragB(Qu + (size_t)q * 512 + d * 16 + hi * 4);

  v8f o[4];
#pragma unroll
  for (int nt = 0; nt < 4; ++nt) o[nt] = vzero8();
  float m = -1e30f, l = 0.f;
  const int nkb = qblk * 4 + 4;
  const u32* VsU = (const u32*)Vs;

  for (int kb = 0; kb < nkb; ++kb) {
    const int k0 = kb * 32;
    {  // stage K block: 32 rows x 32 uints
      int row = tid >> 3, seg = tid & 7;
      *(v4u*)(Ks + row * 32 + seg * 4) =
          *(const v4u*)(Ku + (size_t)(k0 + row) * 512 + seg * 4);
    }
#pragma unroll
    for (int j = 0; j < 4; ++j) {  // stage V transposed: Vs[dv][k]
      int i = tid + j * 256;
      int kk = i >> 5, c = i & 31;
      u32 u = Vu[(size_t)(k0 + kk) * 512 + c];
      Vs[(2 * c) * 32 + kk] = (u16)(u & 0xffffu);
      Vs[(2 * c + 1) * 32 + kk] = (u16)(u >> 16);
    }
    __syncthreads();
    if (k0 <= qw + 15) {   // causal: this wave still has live columns
      v8f s0 = vzero8(), s1 = vzero8();
#pragma unroll
      for (int d = 0; d < 2; ++d) {
        v16bf a0 = ldfragA(Ks + l15 * 32 + d * 16 + hi * 4);
        v16bf a1 = ldfragA(Ks + (16 + l15) * 32 + d * 16 + hi * 4);
        s0 = wmma_bf16(a0, qfrag[d], s0);
        s1 = wmma_bf16(a1, qfrag[d], s1);
      }
      float p0[8], p1[8];
#pragma unroll
      for (int i = 0; i < 8; ++i) { p0[i] = s0[i]; p1[i] = s1[i]; }
      if (k0 + 31 > qw) {   // diagonal block: apply causal mask
        const int kbase = k0 + hi * 8;
#pragma unroll
        for (int i = 0; i < 8; ++i) {
          if (kbase + i > q)      p0[i] = -1e30f;
          if (kbase + 16 + i > q) p1[i] = -1e30f;
        }
      }
      float tmax = -1e30f;
#pragma unroll
      for (int i = 0; i < 8; ++i) tmax = fmaxf(tmax, fmaxf(p0[i], p1[i]));
      tmax = fmaxf(tmax, __shfl_xor(tmax, 16, 32));  // join the two k-halves
      const float mnew = fmaxf(m, tmax);
      const float scale = __expf(m - mnew);
      float tsum = 0.f;
#pragma unroll
      for (int i = 0; i < 8; ++i) {
        p0[i] = __expf(p0[i] - mnew);
        p1[i] = __expf(p1[i] - mnew);
        tsum += p0[i] + p1[i];
      }
      tsum += __shfl_xor(tsum, 16, 32);
      l = l * scale + tsum;
      m = mnew;
      const int qoff = hi * 8;
#pragma unroll
      for (int i = 0; i < 8; ++i) {   // rescale O rows (q = element index)
        float f = __shfl(scale, qoff + i, 32);
#pragma unroll
        for (int nt = 0; nt < 4; ++nt) o[nt][i] *= f;
      }
      v8u pu;   // pack P into A-fragment (C-layout of S^T == A-layout of P)
#pragma unroll
      for (int j2 = 0; j2 < 4; ++j2) {
        pu[j2] = packbf(p0[2 * j2], p0[2 * j2 + 1]);
        pu[4 + j2] = packbf(p1[2 * j2], p1[2 * j2 + 1]);
      }
      v16bf pfrag = __builtin_bit_cast(v16bf, pu);
#pragma unroll
      for (int nt = 0; nt < 4; ++nt) {
        v16bf vb = ldfragB(VsU + (nt * 16 + l15) * 16 + hi * 4);
        o[nt] = wmma_bf16(pfrag, vb, o[nt]);
      }
    }
    __syncthreads();
  }

  const float invl = 1.f / l;
  const int qoff = hi * 8;
  u16* Ob = Og + (size_t)b * TT * 1024 + h * 64;
#pragma unroll
  for (int i = 0; i < 8; ++i) {
    float f = __shfl(invl, qoff + i, 32);
    const size_t rr = (size_t)(qw + qoff + i) * 1024;
#pragma unroll
    for (int nt = 0; nt < 4; ++nt)
      Ob[rr + nt * 16 + l15] = f2bf(o[nt][i] * f);
  }
}

// ---------------------------------------------------------------------------

extern "C" void kernel_launch(void* const* d_in, const int* in_sizes, int n_in,
                              void* d_out, int out_size, void* d_ws,
                              size_t ws_size, hipStream_t stream) {
  (void)in_sizes; (void)n_in; (void)out_size; (void)ws_size;
  const float* x    = (const float*)d_in[0];
  const float* ln1w = (const float*)d_in[1];
  const float* ln1b = (const float*)d_in[2];
  const float* Wq   = (const float*)d_in[3];
  const float* Wk   = (const float*)d_in[4];
  const float* Wv   = (const float*)d_in[5];
  const float* Wo   = (const float*)d_in[6];
  const float* ln2w = (const float*)d_in[7];
  const float* ln2b = (const float*)d_in[8];
  const float* W1   = (const float*)d_in[9];
  const float* W2   = (const float*)d_in[10];

  char* ws = (char*)d_ws;
  const size_t MB = 1024ull * 1024ull;
  u16* Wqt = (u16*)(ws + 0 * MB);    // [1024][1024] bf16
  u16* Wkt = (u16*)(ws + 2 * MB);
  u16* Wvt = (u16*)(ws + 4 * MB);
  u16* Wot = (u16*)(ws + 6 * MB);
  u16* W1t = (u16*)(ws + 8 * MB);    // [4096][1024] bf16
  u16* W2t = (u16*)(ws + 16 * MB);   // [1024][4096] bf16
  u16* H1  = (u16*)(ws + 24 * MB);   // ln1 out, bf16 [4096][1024]
  u16* Qb  = (u16*)(ws + 32 * MB);   // pre-scaled by 1/sqrt(d)
  u16* Kb  = (u16*)(ws + 40 * MB);
  u16* Vb  = (u16*)(ws + 48 * MB);
  u16* Ab  = (u16*)(ws + 56 * MB);   // attention out (heads concat), bf16
  float* X1 = (float*)(ws + 64 * MB);  // x + attn@Wo, f32 [4096][1024]
  u16* H2  = (u16*)(ws + 80 * MB);   // ln2 out, bf16
  u16* G   = (u16*)(ws + 88 * MB);   // gelu(h2@W1), bf16 [4096][4096]

  // 1) weights -> bf16, transposed (B-fragment reads become contiguous)
  k_transpose_bf16<<<(DMODEL * DMODEL) / 256, 256, 0, stream>>>(Wq, Wqt, DMODEL, DMODEL);
  k_transpose_bf16<<<(DMODEL * DMODEL) / 256, 256, 0, stream>>>(Wk, Wkt, DMODEL, DMODEL);
  k_transpose_bf16<<<(DMODEL * DMODEL) / 256, 256, 0, stream>>>(Wv, Wvt, DMODEL, DMODEL);
  k_transpose_bf16<<<(DMODEL * DMODEL) / 256, 256, 0, stream>>>(Wo, Wot, DMODEL, DMODEL);
  k_transpose_bf16<<<(DMODEL * DFFN) / 256, 256, 0, stream>>>(W1, W1t, DMODEL, DFFN);
  k_transpose_bf16<<<(DFFN * DMODEL) / 256, 256, 0, stream>>>(W2, W2t, DFFN, DMODEL);

  // 2) ln1
  k_layernorm_bf16<<<MTOK, 256, 0, stream>>>(x, ln1w, ln1b, H1);

  // 3) QKV projections (1/sqrt(64) folded into Q)
  dim3 g1(DMODEL / 256, MTOK / 64);
  k_gemm_bf16<false><<<g1, 256, 0, stream>>>(H1, Wqt, MTOK, DMODEL, DMODEL, 0.125f, nullptr, nullptr, Qb);
  k_gemm_bf16<false><<<g1, 256, 0, stream>>>(H1, Wkt, MTOK, DMODEL, DMODEL, 1.0f, nullptr, nullptr, Kb);
  k_gemm_bf16<false><<<g1, 256, 0, stream>>>(H1, Wvt, MTOK, DMODEL, DMODEL, 1.0f, nullptr, nullptr, Vb);

  // 4) causal flash attention
  dim3 ga(TT / 128, BB * NHEADS);
  k_attention<<<ga, 256, 0, stream>>>(Qb, Kb, Vb, Ab);

  // 5) output projection + residual -> X1 (f32)
  k_gemm_bf16<false><<<g1, 256, 0, stream>>>(Ab, Wot, MTOK, DMODEL, DMODEL, 1.0f, x, X1, nullptr);

  // 6) ln2
  k_layernorm_bf16<<<MTOK, 256, 0, stream>>>(X1, ln2w, ln2b, H2);

  // 7) FFN up + exact-erf GELU
  dim3 g2(DFFN / 256, MTOK / 64);
  k_gemm_bf16<true><<<g2, 256, 0, stream>>>(H2, W1t, MTOK, DFFN, DMODEL, 1.0f, nullptr, nullptr, G);

  // 8) FFN down + residual -> d_out (f32)
  k_gemm_bf16<false><<<g1, 256, 0, stream>>>(G, W2t, MTOK, DMODEL, DFFN, 1.0f, X1, (float*)d_out, nullptr);
}
